// GraphEncoder_18210661335049
// MI455X (gfx1250) — compile-verified, matrix operand order checked by reading, not run on hardware
//
#include <hip/hip_runtime.h>
#include <hip/hip_bf16.h>

// ---------------------------------------------------------------------------
// 2-layer GAT for MI455X (gfx1250, wave32).
//  * Projections h = in @ W^T via v_wmma_f32_16x16x32_f16 (f16 A/B, f32 acc),
//    K templated so the k-loop fully unrolls into back-to-back WMMAs.
//  * Edge softmax + aggregation via 3 streaming passes with L2-resident
//    f32 atomics (all per-node state fits in the 192MB L2).
//  * Bias folded into accumulator init; ReLU in place; block-per-channel mean.
// ---------------------------------------------------------------------------

typedef __attribute__((ext_vector_type(16))) _Float16 v16h;
typedef __attribute__((ext_vector_type(8)))  float    v8f;

#define N_NODES   100000
#define N_EDGES   1600000
#define N_TOT     (N_EDGES + N_NODES)   // edges + self loops
#define IN_DIM    128
#define HID       64
#define NEG_SLOPE 0.2f

// workspace layout (in floats)
#define O_H     0                              // [N,64] projected features
#define O_AGG   (O_H     + N_NODES * 64)       // [N,64] layer-1 accum / layer-2 input
#define O_AGG2  (O_AGG   + N_NODES * 64)       // [N,64] layer-2 accum
#define O_ASRC  (O_AGG2  + N_NODES * 64)       // [N] alpha_src
#define O_ADST  (O_ASRC  + N_NODES)            // [N] alpha_dst
#define O_M     (O_ADST  + N_NODES)            // [N] segment max
#define O_DEN   (O_M     + N_NODES)            // [N] segment sum
#define O_LOGIT (O_DEN   + N_NODES)            // [E+N] logits, then exp()

// ---------------------------------------------------------------------------
// init per-layer state: accumulator = bias (reference adds b after aggregation),
// segment max = -inf, denom = 0.
__global__ void k_init(float* __restrict__ agg, float* __restrict__ mmax,
                       float* __restrict__ den, const float* __restrict__ bias) {
  int t = blockIdx.x * blockDim.x + threadIdx.x;
  if (t < N_NODES * 64) {
    int c = t & 63;
    int node = t >> 6;
    agg[t] = bias[c];
    if (c == 0) { mmax[node] = -__builtin_inff(); den[node] = 0.0f; }
  }
}

// ---------------------------------------------------------------------------
// H[M,64] = X[M,K] @ W[64,K]^T  via WMMA f32_16x16x32_f16, K compile-time.
// Block = 128 threads = 4 waves; wave w owns output columns [16w,16w+16).
// Per-lane A/B element mapping per CDNA5 ISA 16-bit 16x32 layout:
//   lane = (half<<4)|mn ; v16h element e (VGPR j=e>>1, sub=e&1) holds
//   k = (j&3)*2 + sub + half*8 + (j>>2)*16.
template <int K>
__global__ __launch_bounds__(128) void k_gemm_f16(const float* __restrict__ X,
                                                  const float* __restrict__ W,
                                                  float* __restrict__ H) {
  int lane = threadIdx.x & 31;
  int wv   = threadIdx.x >> 5;      // 0..3 -> output column tile
  int row0 = blockIdx.x * 16;
  int n0   = wv * 16;
  int half = lane >> 4;
  int mn   = lane & 15;

  const float* xrow = X + (size_t)(row0 + mn) * K;  // A: row (row0+mn)
  const float* wrow = W + (size_t)(n0 + mn) * K;    // B: col (n0+mn) == W row

  v8f acc = {};
#pragma unroll
  for (int k0 = 0; k0 < K; k0 += 32) {
    v16h a, b;
#pragma unroll
    for (int j = 0; j < 8; ++j) {
      int kk = ((j & 3) * 2) + half * 8 + ((j >> 2) * 16) + k0;  // even -> 8B aligned
      float2 xa = *(const float2*)(xrow + kk);
      float2 wb = *(const float2*)(wrow + kk);
      a[2 * j]     = (_Float16)xa.x;
      a[2 * j + 1] = (_Float16)xa.y;
      b[2 * j]     = (_Float16)wb.x;
      b[2 * j + 1] = (_Float16)wb.y;
    }
    acc = __builtin_amdgcn_wmma_f32_16x16x32_f16(
        /*neg_a=*/false, a, /*neg_b=*/false, b,
        /*c_mod=*/(short)0, acc, /*reuse_a=*/false, /*reuse_b=*/false);
  }
#pragma unroll
  for (int r = 0; r < 8; ++r)       // D: VGPR r -> row r + 8*half, col mn
    H[(size_t)(row0 + r + half * 8) * 64 + n0 + mn] = acc[r];
}

// ---------------------------------------------------------------------------
// per-node attention scalars: asrc = h . a_src , adst = h . a_dst
__global__ void k_alpha(const float* __restrict__ H, const float* __restrict__ av_s,
                        const float* __restrict__ av_d, float* __restrict__ as_,
                        float* __restrict__ ad_) {
  int node = blockIdx.x * blockDim.x + threadIdx.x;
  if (node < N_NODES) {
    const float* hr = H + (size_t)node * 64;
    float s = 0.0f, d = 0.0f;
#pragma unroll
    for (int c = 0; c < 64; ++c) {
      float v = hr[c];
      s += v * av_s[c];
      d += v * av_d[c];
    }
    as_[node] = s;
    ad_[node] = d;
  }
}

// sign-magnitude trick: works for mixed signs when init = -inf
__device__ __forceinline__ void atomic_max_f32(float* a, float v) {
  if (v >= 0.0f) atomicMax((int*)a, __float_as_int(v));
  else           atomicMin((unsigned int*)a, __float_as_uint(v));
}

// pass 1: logits + segment max over incoming edges
__global__ void k_edge_max(const int* __restrict__ ei, const float* __restrict__ as_,
                           const float* __restrict__ ad_, float* __restrict__ logit,
                           float* __restrict__ mmax) {
  int e = blockIdx.x * blockDim.x + threadIdx.x;
  if (e < N_TOT) {
    __builtin_prefetch(ei + e + 4096, 0, 0);  // global_prefetch_b8 on index stream
    int s, d;
    if (e < N_EDGES) { s = ei[e]; d = ei[N_EDGES + e]; }
    else             { s = d = e - N_EDGES; }
    float x = as_[s] + ad_[d];
    float l = x > 0.0f ? x : NEG_SLOPE * x;   // leaky_relu(0.2)
    logit[e] = l;
    atomic_max_f32(mmax + d, l);
  }
}

// pass 2: ex = exp(logit - max[dst]); denom[dst] += ex; store ex in place
__global__ void k_edge_exp(const int* __restrict__ ei, const float* __restrict__ mmax,
                           float* __restrict__ logit, float* __restrict__ den) {
  int e = blockIdx.x * blockDim.x + threadIdx.x;
  if (e < N_TOT) {
    int d;
    if (e < N_EDGES) d = ei[N_EDGES + e];
    else             d = e - N_EDGES;
    float ex = expf(logit[e] - mmax[d]);
    logit[e] = ex;
    atomicAdd(den + d, ex);
  }
}

// pass 3: agg[dst,:] += (ex/denom[dst]) * h[src,:]   (64 threads per edge)
__global__ void k_edge_agg(const int* __restrict__ ei, const float* __restrict__ logit,
                           const float* __restrict__ den, const float* __restrict__ H,
                           float* __restrict__ agg) {
  long long t = (long long)blockIdx.x * blockDim.x + threadIdx.x;
  if (t < (long long)N_TOT * 64) {
    int e = (int)(t >> 6);
    int c = (int)(t & 63);
    int s, d;
    if (e < N_EDGES) { s = ei[e]; d = ei[N_EDGES + e]; }
    else             { s = d = e - N_EDGES; }
    float alpha = logit[e] / den[d];
    atomicAdd(agg + (size_t)d * 64 + c, H[(size_t)s * 64 + c] * alpha);
  }
}

__global__ void k_relu(float* __restrict__ a, int n) {
  int t = blockIdx.x * blockDim.x + threadIdx.x;
  if (t < n) a[t] = fmaxf(a[t], 0.0f);
}

// mean over nodes: one block per channel (64 blocks)
__global__ void k_mean(const float* __restrict__ agg2, float* __restrict__ out) {
  __shared__ float red[256];
  int c = blockIdx.x;
  float s = 0.0f;
  for (int node = threadIdx.x; node < N_NODES; node += blockDim.x)
    s += agg2[(size_t)node * 64 + c];
  red[threadIdx.x] = s;
  __syncthreads();
  for (int w = 128; w > 0; w >>= 1) {
    if ((int)threadIdx.x < w) red[threadIdx.x] += red[threadIdx.x + w];
    __syncthreads();
  }
  if (threadIdx.x == 0) out[c] = red[0] / (float)N_NODES;
}

// ---------------------------------------------------------------------------
extern "C" void kernel_launch(void* const* d_in, const int* in_sizes, int n_in,
                              void* d_out, int out_size, void* d_ws, size_t ws_size,
                              hipStream_t stream) {
  const float* x   = (const float*)d_in[0];
  const int*   ei  = (const int*)d_in[1];
  // d_in[2] edge_attr unused (edge_dim=None in reference)
  const float* W1  = (const float*)d_in[3];
  const float* as1 = (const float*)d_in[4];
  const float* ad1 = (const float*)d_in[5];
  const float* b1  = (const float*)d_in[6];
  const float* W2  = (const float*)d_in[7];
  const float* as2 = (const float*)d_in[8];
  const float* ad2 = (const float*)d_in[9];
  const float* b2  = (const float*)d_in[10];

  float* ws    = (float*)d_ws;
  float* H     = ws + O_H;
  float* AGG   = ws + O_AGG;
  float* AGG2  = ws + O_AGG2;
  float* ASRC  = ws + O_ASRC;
  float* ADST  = ws + O_ADST;
  float* MMAX  = ws + O_M;
  float* DEN   = ws + O_DEN;
  float* LOGIT = ws + O_LOGIT;
  float* out   = (float*)d_out;

  const int gN64  = (N_NODES * 64 + 255) / 256;
  const int gNode = (N_NODES + 255) / 256;
  const int gE    = (N_TOT + 255) / 256;
  const int gEC   = (int)(((long long)N_TOT * 64 + 255) / 256);
  const int gGemm = N_NODES / 16;   // 100000 = 6250 * 16, exact

  // ---- layer 1: x[100k,128] -> 64 ----
  k_init            <<<gN64, 256, 0, stream>>>(AGG, MMAX, DEN, b1);
  k_gemm_f16<IN_DIM><<<gGemm, 128, 0, stream>>>(x, W1, H);
  k_alpha           <<<gNode, 256, 0, stream>>>(H, as1, ad1, ASRC, ADST);
  k_edge_max        <<<gE, 256, 0, stream>>>(ei, ASRC, ADST, LOGIT, MMAX);
  k_edge_exp        <<<gE, 256, 0, stream>>>(ei, MMAX, LOGIT, DEN);
  k_edge_agg        <<<gEC, 256, 0, stream>>>(ei, LOGIT, DEN, H, AGG);
  k_relu            <<<gN64, 256, 0, stream>>>(AGG, N_NODES * 64);

  // ---- layer 2: relu(h1)[100k,64] -> 64 ----
  k_init            <<<gN64, 256, 0, stream>>>(AGG2, MMAX, DEN, b2);
  k_gemm_f16<HID>   <<<gGemm, 128, 0, stream>>>(AGG, W2, H);
  k_alpha           <<<gNode, 256, 0, stream>>>(H, as2, ad2, ASRC, ADST);
  k_edge_max        <<<gE, 256, 0, stream>>>(ei, ASRC, ADST, LOGIT, MMAX);
  k_edge_exp        <<<gE, 256, 0, stream>>>(ei, MMAX, LOGIT, DEN);
  k_edge_agg        <<<gEC, 256, 0, stream>>>(ei, LOGIT, DEN, H, AGG2);

  // ---- global mean pool -> [1,64] ----
  k_mean<<<64, 256, 0, stream>>>(AGG2, out);
}
